// CrossSparseAggrNet_979252544027
// MI455X (gfx1250) — matrix operand order repaired and checked by reading, not compiled, not run on hardware
//
#include <hip/hip_runtime.h>
#include <math.h>

#define BV 48
#define BT 48
#define MW 32
#define NTOK 196
#define NTOKP 208   // padded token count (13 x 16), rows 196..207 are zero
#define CDIM 512
#define HIDN 102
#define KEEPED 39
#define NKEEP 98
#define NTOKS 41    // 1 cls + 39 aggr + 1 extra

typedef __bf16 bf16;
typedef __attribute__((ext_vector_type(16))) __bf16 v16bf;
typedef __attribute__((ext_vector_type(8)))  float  v8f;

static __device__ __forceinline__ v8f wmma_bf16(v16bf a, v16bf b, v8f c) {
  return __builtin_amdgcn_wmma_f32_16x16x32_bf16(false, a, false, b, (short)0, c, false, false);
}

static __device__ __forceinline__ float wave_sum(float v) {
#pragma unroll
  for (int o = 16; o > 0; o >>= 1) v += __shfl_xor(v, o, 32);
  return v;
}
static __device__ __forceinline__ float wave_max(float v) {
#pragma unroll
  for (int o = 16; o > 0; o >>= 1) v = fmaxf(v, __shfl_xor(v, o, 32));
  return v;
}
static __device__ __forceinline__ float block_sum8(float v, volatile float* red, int tid) {
  float w = wave_sum(v);
  if ((tid & 31) == 0) red[tid >> 5] = w;
  __syncthreads();
  float tot = red[0] + red[1] + red[2] + red[3] + red[4] + red[5] + red[6] + red[7];
  __syncthreads();
  return tot;
}

// A fragment (16x32 bf16 tile of row-major M x ldk matrix at (m0,k0)).
// Lane l: row m0+(l&15); K values 16*hi..16*hi+15 contiguous, hi=l>>4.
static __device__ __forceinline__ v16bf load_a_frag(const bf16* A, int ldk, int m0, int k0, int lane) {
  int m  = m0 + (lane & 15);
  int hi = lane >> 4;
  const bf16* p = A + (size_t)m * ldk + k0 + 16 * hi;
  v16bf r;
#pragma unroll
  for (int h = 0; h < 16; ++h) r[h] = p[h];
  return r;
}
// B fragment (32x16 tile) from transposed storage Bt (row-major N x ldk, Bt[n][k]==B[k][n]).
// Lane l: column n0+(l&15); K values 16*hi..16*hi+15 contiguous.
static __device__ __forceinline__ v16bf load_bt_frag(const bf16* Bt, int ldk, int n0, int k0, int lane) {
  int n  = n0 + (lane & 15);
  int hi = lane >> 4;
  const bf16* p = Bt + (size_t)n * ldk + k0 + 16 * hi;
  v16bf r;
#pragma unroll
  for (int h = 0; h < 16; ++h) r[h] = p[h];
  return r;
}

// ------------------------------------------------------------------ prep kernels

__global__ __launch_bounds__(256)
void prep_image_kernel(const float* __restrict__ img,  // (48,197,512)
                       bf16* __restrict__ spat_bf,     // (48,208,512) l2-normalized spatial, zero-padded
                       bf16* __restrict__ cls_bf,      // (48,512)     l2-normalized cls
                       float* __restrict__ s_im)       // (48,196)
{
  __shared__ float meanv[CDIM];
  __shared__ float red[8];
  int b = blockIdx.x, tid = threadIdx.x, lane = tid & 31, wave = tid >> 5;
  const float* base = img + (size_t)b * (NTOK + 1) * CDIM;

  for (int c = tid; c < CDIM; c += blockDim.x) {
    float s = 0.f;
    for (int n = 0; n < NTOK; ++n) s += base[(size_t)(n + 1) * CDIM + c];
    meanv[c] = s * (1.0f / NTOK);
  }
  __syncthreads();
  float part = 0.f;
  for (int c = tid; c < CDIM; c += blockDim.x) { float v = meanv[c]; part += v * v; }
  float tot  = block_sum8(part, red, tid);
  float ginv = 1.0f / fmaxf(sqrtf(tot), 1e-12f);

  for (int t = wave; t < NTOK + 1; t += 8) {
    const float* row = base + (size_t)t * CDIM;
    float ss = 0.f;
    for (int c = lane; c < CDIM; c += 32) { float v = row[c]; ss += v * v; }
    ss = wave_sum(ss);
    float invn = 1.0f / fmaxf(sqrtf(ss), 1e-12f);
    if (t == 0) {
      for (int c = lane; c < CDIM; c += 32) cls_bf[(size_t)b * CDIM + c] = (bf16)(row[c] * invn);
    } else {
      int n = t - 1;
      float dot = 0.f;
      for (int c = lane; c < CDIM; c += 32) {
        float sn = row[c] * invn;
        spat_bf[((size_t)b * NTOKP + n) * CDIM + c] = (bf16)sn;
        dot += meanv[c] * ginv * sn;
      }
      dot = wave_sum(dot);
      if (lane == 0) s_im[b * NTOK + n] = dot;
    }
  }
  // zero the 12 padding token rows (makes phase-1 B loads unguarded)
  for (int n = NTOK + wave; n < NTOKP; n += 8)
    for (int c = lane; c < CDIM; c += 32)
      spat_bf[((size_t)b * NTOKP + n) * CDIM + c] = (bf16)0.0f;
}

__global__ __launch_bounds__(256)
void prep_caption_kernel(const float* __restrict__ cap, bf16* __restrict__ cap_bf)
{
  int jj = blockIdx.x, lane = threadIdx.x & 31, wave = threadIdx.x >> 5;
  for (int m = wave; m < MW; m += 8) {
    const float* row = cap + ((size_t)jj * MW + m) * CDIM;
    float ss = 0.f;
    for (int c = lane; c < CDIM; c += 32) { float v = row[c]; ss += v * v; }
    ss = wave_sum(ss);
    float inv = 1.0f / fmaxf(sqrtf(ss), 1e-12f);
    for (int c = lane; c < CDIM; c += 32)
      cap_bf[((size_t)jj * MW + m) * CDIM + c] = (bf16)(row[c] * inv);
  }
}

__global__ __launch_bounds__(256)
void prep_weights_kernel(const float* __restrict__ w1, const float* __restrict__ w2,
                         bf16* __restrict__ w1t,   // (112,512) = w1^T padded
                         bf16* __restrict__ w2t)   // (48,128)  = w2^T padded
{
  int idx = blockIdx.x * blockDim.x + threadIdx.x;
  int stride = gridDim.x * blockDim.x;
  const int n1 = 112 * CDIM;
  for (int i = idx; i < n1; i += stride) {
    int n = i / CDIM, k = i % CDIM;
    float v = (n < HIDN) ? w1[(size_t)k * HIDN + n] : 0.f;
    w1t[(size_t)n * CDIM + k] = (bf16)v;
  }
  const int n2 = 48 * 128;
  for (int i = idx; i < n2; i += stride) {
    int n = i / 128, k = i % 128;
    float v = (n < KEEPED && k < HIDN) ? w2[(size_t)k * KEEPED + n] : 0.f;
    w2t[(size_t)n * 128 + k] = (bf16)v;
  }
}

// ------------------------------------------------------------------ pair kernel

// dynamic LDS layout (bytes)
#define SELT_OFF   0u        // bf16 [512][128]  sel^T (raw kept tokens); later bf16 toks[48][512]
#define SELLN_OFF  131072u   // bf16 [112][512]  LN(sel); later f32 aggr[48][512]
#define H1_OFF     245760u   // bf16 [112][128]  gelu(LN@w1)
#define H2T_OFF    274432u   // f32  [48][112]   (h@w2)^T * scale; later f32 c2t[32][48]
#define WB_OFF     296960u   // bf16 [48][128]   softmax weights
#define SCORE_OFF  309248u   // f32  [256]
#define SIDX_OFF   310272u   // i32  [256]
#define EXTRA_OFF  311296u   // f32  [512]
#define NONW_OFF   313344u   // f32  [128]
#define RED_OFF    313856u   // f32  [128]
#define SMEM_TOTAL 314368u

__global__ __launch_bounds__(256)
void pair_kernel(const float* __restrict__ img,     // (48,197,512) raw
                 const float* __restrict__ ln_g,
                 const float* __restrict__ ln_b,
                 const float* __restrict__ b1,
                 const float* __restrict__ b2,
                 const float* __restrict__ scale,
                 const bf16* __restrict__ spat_bf,
                 const bf16* __restrict__ cls_bf,
                 const float* __restrict__ s_im,
                 const bf16* __restrict__ cap_bf,
                 const bf16* __restrict__ w1t,
                 const bf16* __restrict__ w2t,
                 float* __restrict__ out)            // (48,48) = sims.T
{
  extern __shared__ char smem[];
  const int b = blockIdx.x, j = blockIdx.y;
  const int tid = threadIdx.x, lane = tid & 31, wave = tid >> 5;

  bf16*  selT  = (bf16*)(smem + SELT_OFF);
  bf16*  selLN = (bf16*)(smem + SELLN_OFF);
  bf16*  h1    = (bf16*)(smem + H1_OFF);
  float* h2T   = (float*)(smem + H2T_OFF);
  bf16*  wbf   = (bf16*)(smem + WB_OFF);
  float* score = (float*)(smem + SCORE_OFF);
  int*   sidx  = (int*)(smem + SIDX_OFF);
  float* extra = (float*)(smem + EXTRA_OFF);
  float* nonw  = (float*)(smem + NONW_OFF);
  float* red   = (float*)(smem + RED_OFF);
  float* aggr  = (float*)(smem + SELLN_OFF);  // alias (selLN dead after MLP)
  bf16*  toks  = (bf16*)(smem + SELT_OFF);    // alias (selT dead after aggr)
  float* c2t   = (float*)(smem + H2T_OFF);    // alias (h2T dead after softmax)

  // phase 0: zero all matmul staging (zero padding is load-bearing)
  for (unsigned off = tid; off * 4u < SCORE_OFF; off += blockDim.x)
    ((float*)smem)[off] = 0.0f;
  score[tid] = -3.0e38f;
  sidx[tid]  = tid;
  __syncthreads();

  const bf16* capj  = cap_bf + (size_t)j * MW * CDIM;
  const bf16* spatb = spat_bf + (size_t)b * NTOKP * CDIM;
  const float* imgb = img + (size_t)b * (NTOK + 1) * CDIM;

  // phase 1: c2i = cap(32x512) @ spat^T -> column max -> score
  for (int nt = wave; nt < 13; nt += 8) {
    int n0 = nt * 16;
    v8f acc0 = {}; v8f acc1 = {};
    for (int k0 = 0; k0 < CDIM; k0 += 32) {
      v16bf bb = load_bt_frag(spatb, CDIM, n0, k0, lane);   // rows 196..207 are zero
      v16bf a0 = load_a_frag(capj, CDIM, 0,  k0, lane);
      v16bf a1 = load_a_frag(capj, CDIM, 16, k0, lane);
      acc0 = wmma_bf16(a0, bb, acc0);
      acc1 = wmma_bf16(a1, bb, acc1);
    }
    float mx = -3.0e38f;
#pragma unroll
    for (int i = 0; i < 8; ++i) mx = fmaxf(mx, fmaxf(acc0[i], acc1[i]));
    mx = fmaxf(mx, __shfl_xor(mx, 16, 32));
    int n = n0 + (lane & 15);
    if (lane < 16 && n < NTOK) score[n] = s_im[b * NTOK + n] + mx;
  }
  __syncthreads();

  // phase 2: bitonic sort 256 (descending score, carry index)
  for (int k = 2; k <= 256; k <<= 1) {
    for (int s = k >> 1; s > 0; s >>= 1) {
      int i = tid, ixj = i ^ s;
      if (ixj > i) {
        float va = score[i], vb = score[ixj];
        bool desc = ((i & k) == 0);
        if (desc ? (va < vb) : (va > vb)) {
          score[i] = vb; score[ixj] = va;
          int tt = sidx[i]; sidx[i] = sidx[ixj]; sidx[ixj] = tt;
        }
      }
      __syncthreads();
    }
  }

  // phase 3: softmax over tail (sorted desc -> max is score[98]) and "extra" token
  float maxv = score[NKEEP];
  float part = 0.f;
  for (int p = NKEEP + tid; p < NTOK; p += blockDim.x) part += expf(score[p] - maxv);
  float ssum = block_sum8(part, red, tid);
  float invs = 1.0f / ssum;
  for (int p = NKEEP + tid; p < NTOK; p += blockDim.x) nonw[p - NKEEP] = expf(score[p] - maxv) * invs;
  __syncthreads();
  for (int c = tid; c < CDIM; c += blockDim.x) {
    float acc = 0.f;
    for (int p = 0; p < NTOK - NKEEP; ++p) {
      int idx = sidx[NKEEP + p];
      acc += nonw[p] * imgb[(size_t)(1 + idx) * CDIM + c];
    }
    extra[c] = acc;
  }

  // phase 4: kept tokens -> selT (raw bf16, transposed) and selLN (LayerNorm bf16)
  for (int t = wave; t < NKEEP; t += 8) {
    int src = sidx[t];
    const float* row = imgb + (size_t)(1 + src) * CDIM;
    float vals[16];
    float s1 = 0.f, s2 = 0.f;
#pragma unroll
    for (int u = 0; u < 16; ++u) {
      float v = row[lane + u * 32];
      vals[u] = v; s1 += v; s2 += v * v;
    }
    s1 = wave_sum(s1); s2 = wave_sum(s2);
    float mu   = s1 * (1.0f / CDIM);
    float var  = s2 * (1.0f / CDIM) - mu * mu;
    float rstd = rsqrtf(var + 1e-5f);
#pragma unroll
    for (int u = 0; u < 16; ++u) {
      int c = lane + u * 32;
      float x = vals[u];
      selT[(size_t)c * 128 + t]   = (bf16)x;
      selLN[(size_t)t * CDIM + c] = (bf16)((x - mu) * rstd * ln_g[c] + ln_b[c]);
    }
  }
  __syncthreads();

  // phase 5: h1 = gelu(selLN(112x512) @ w1(512x112) + b1)
  for (int tt = wave; tt < 49; tt += 8) {
    int m0 = (tt / 7) * 16, n0 = (tt % 7) * 16;
    v8f acc = {};
    for (int k0 = 0; k0 < CDIM; k0 += 32) {
      v16bf a  = load_a_frag(selLN, CDIM, m0, k0, lane);
      v16bf bb = load_bt_frag(w1t, CDIM, n0, k0, lane);
      acc = wmma_bf16(a, bb, acc);
    }
    int coln  = n0 + (lane & 15);
    int rbase = (lane >> 4) * 8;
    float bias = (coln < HIDN) ? b1[coln] : 0.f;
#pragma unroll
    for (int i = 0; i < 8; ++i) {
      float x = acc[i] + bias;
      float g = 0.5f * x * (1.0f + erff(x * 0.70710678118f));
      h1[(size_t)(m0 + rbase + i) * 128 + coln] = (bf16)g;
    }
  }
  __syncthreads();

  // phase 6: h2 = h1(112x128) @ w2(128x48) + b2 ; store transposed * scale
  float sc = scale[0];
  for (int tt = wave; tt < 21; tt += 8) {
    int m0 = (tt / 3) * 16, n0 = (tt % 3) * 16;
    v8f acc = {};
    for (int k0 = 0; k0 < 128; k0 += 32) {
      v16bf a  = load_a_frag(h1, 128, m0, k0, lane);
      v16bf bb = load_bt_frag(w2t, 128, n0, k0, lane);
      acc = wmma_bf16(a, bb, acc);
    }
    int coln  = n0 + (lane & 15);
    int rbase = (lane >> 4) * 8;
    float bias = (coln < KEEPED) ? b2[coln] : 0.f;
#pragma unroll
    for (int i = 0; i < 8; ++i)
      h2T[(size_t)coln * 112 + (m0 + rbase + i)] = (acc[i] + bias) * sc;
  }
  __syncthreads();

  // phase 7a: per-k-slot softmax over 98 tokens -> wbf
  for (int r = wave; r < KEEPED; r += 8) {
    const float* rp = h2T + (size_t)r * 112;
    float e[4];
    float mx = -3.0e38f;
#pragma unroll
    for (int u = 0; u < 4; ++u) {
      int t = lane + u * 32;
      float v = (t < NKEEP) ? rp[t] : -3.0e38f;
      e[u] = v; mx = fmaxf(mx, v);
    }
    mx = wave_max(mx);
    float sum = 0.f;
#pragma unroll
    for (int u = 0; u < 4; ++u) {
      int t = lane + u * 32;
      float ev = (t < NKEEP) ? expf(e[u] - mx) : 0.f;
      e[u] = ev; sum += ev;
    }
    sum = wave_sum(sum);
    float inv = 1.0f / sum;
#pragma unroll
    for (int u = 0; u < 4; ++u) {
      int t = lane + u * 32;
      if (t < NKEEP) wbf[(size_t)r * 128 + t] = (bf16)(e[u] * inv);
    }
  }
  __syncthreads();

  // phase 7b: aggr = w(48x128) @ sel(128x512)
  for (int tt = wave; tt < 96; tt += 8) {
    int m0 = (tt / 32) * 16, n0 = (tt % 32) * 16;
    v8f acc = {};
    for (int k0 = 0; k0 < 128; k0 += 32) {
      v16bf a  = load_a_frag(wbf, 128, m0, k0, lane);
      v16bf bb = load_bt_frag(selT, 128, n0, k0, lane);
      acc = wmma_bf16(a, bb, acc);
    }
    int coln  = n0 + (lane & 15);
    int rbase = (lane >> 4) * 8;
#pragma unroll
    for (int i = 0; i < 8; ++i)
      aggr[(size_t)(m0 + rbase + i) * CDIM + coln] = acc[i];
  }
  __syncthreads();

  // phase 7c: toks = l2norm rows of [cls, aggr(39), extra], padded to 48
  for (int r = wave; r < 48; r += 8) {
    if (r == 0) {
      for (int c = lane; c < CDIM; c += 32) toks[c] = cls_bf[(size_t)b * CDIM + c];
    } else if (r <= KEEPED) {
      const float* ap = aggr + (size_t)(r - 1) * CDIM;
      float ss = 0.f;
      for (int c = lane; c < CDIM; c += 32) { float v = ap[c]; ss += v * v; }
      ss = wave_sum(ss);
      float inv = 1.0f / fmaxf(sqrtf(ss), 1e-12f);
      for (int c = lane; c < CDIM; c += 32) toks[(size_t)r * CDIM + c] = (bf16)(ap[c] * inv);
    } else if (r == KEEPED + 1) {
      float ss = 0.f;
      for (int c = lane; c < CDIM; c += 32) { float v = extra[c]; ss += v * v; }
      ss = wave_sum(ss);
      float inv = 1.0f / fmaxf(sqrtf(ss), 1e-12f);
      for (int c = lane; c < CDIM; c += 32) toks[(size_t)r * CDIM + c] = (bf16)(extra[c] * inv);
    } else {
      for (int c = lane; c < CDIM; c += 32) toks[(size_t)r * CDIM + c] = (bf16)0.0f;
    }
  }
  __syncthreads();

  // phase 8: c2t = cap(32x512) @ toks^T(512x48), leaky relu
  if (wave < 6) {
    int m0 = (wave / 3) * 16, n0 = (wave % 3) * 16;
    v8f acc = {};
    for (int k0 = 0; k0 < CDIM; k0 += 32) {
      v16bf a  = load_a_frag(capj, CDIM, m0, k0, lane);
      v16bf bb = load_bt_frag(toks, CDIM, n0, k0, lane);
      acc = wmma_bf16(a, bb, acc);
    }
    int coln  = n0 + (lane & 15);
    int rbase = (lane >> 4) * 8;
#pragma unroll
    for (int i = 0; i < 8; ++i) {
      float v = acc[i];
      v = (v >= 0.f) ? v : 0.1f * v;
      c2t[(size_t)(m0 + rbase + i) * 48 + coln] = v;
    }
  }
  __syncthreads();

  // final: row = mean_m max_t, col = mean_t max_m
  if (tid < MW) {
    float mx = -3.0e38f;
    for (int t = 0; t < NTOKS; ++t) mx = fmaxf(mx, c2t[(size_t)tid * 48 + t]);
    red[tid] = mx;
  } else if (tid < MW + NTOKS) {
    int t = tid - MW;
    float mx = -3.0e38f;
    for (int m = 0; m < MW; ++m) mx = fmaxf(mx, c2t[(size_t)m * 48 + t]);
    red[32 + t] = mx;
  }
  __syncthreads();
  if (tid == 0) {
    float rsum = 0.f; for (int m = 0; m < MW; ++m) rsum += red[m];
    float csum = 0.f; for (int t = 0; t < NTOKS; ++t) csum += red[32 + t];
    out[b * BT + j] = rsum * (1.0f / MW) + csum * (1.0f / NTOKS);
  }
}

// ------------------------------------------------------------------ launch

extern "C" void kernel_launch(void* const* d_in, const int* in_sizes, int n_in,
                              void* d_out, int out_size, void* d_ws, size_t ws_size,
                              hipStream_t stream) {
  (void)in_sizes; (void)n_in; (void)out_size; (void)ws_size;
  const float* img   = (const float*)d_in[0];
  const float* cap   = (const float*)d_in[1];
  // d_in[2] = cap_lens (unused by the reference math)
  const float* ln_g  = (const float*)d_in[3];
  const float* ln_b  = (const float*)d_in[4];
  const float* w1    = (const float*)d_in[5];
  const float* b1    = (const float*)d_in[6];
  const float* w2    = (const float*)d_in[7];
  const float* b2    = (const float*)d_in[8];
  const float* scale = (const float*)d_in[9];
  float* out = (float*)d_out;

  char* ws = (char*)d_ws;
  size_t off = 0;
  bf16*  spat_bf = (bf16*)(ws + off);  off += (size_t)BV * NTOKP * CDIM * 2;
  bf16*  cls_bf  = (bf16*)(ws + off);  off += (size_t)BV * CDIM * 2;
  float* s_im    = (float*)(ws + off); off += (size_t)BV * NTOK * 4;
  bf16*  cap_bf  = (bf16*)(ws + off);  off += (size_t)BT * MW * CDIM * 2;
  bf16*  w1t     = (bf16*)(ws + off);  off += (size_t)112 * CDIM * 2;
  bf16*  w2t     = (bf16*)(ws + off);  off += (size_t)48 * 128 * 2;

  prep_image_kernel<<<BV, 256, 0, stream>>>(img, spat_bf, cls_bf, s_im);
  prep_caption_kernel<<<BT, 256, 0, stream>>>(cap, cap_bf);
  prep_weights_kernel<<<64, 256, 0, stream>>>(w1, w2, w1t, w2t);

  hipFuncSetAttribute(reinterpret_cast<const void*>(&pair_kernel),
                      hipFuncAttributeMaxDynamicSharedMemorySize, (int)SMEM_TOTAL);
  pair_kernel<<<dim3(BV, BT), 256, SMEM_TOTAL, stream>>>(
      img, ln_g, ln_b, b1, b2, scale,
      spat_bf, cls_bf, s_im, cap_bf, w1t, w2t, out);
}
